// DSMIL_558345749023
// MI455X (gfx1250) — matrix operand order repaired and checked by reading, not compiled, not run on hardware
//
#include <hip/hip_runtime.h>
#include <math.h>

#define BB 16
#define NN 8192
#define INF_ 2048
#define META 3
#define DQ 128
#define RSTRIDE (INF_ + META) /* 2051 */

typedef __attribute__((ext_vector_type(16))) __bf16 v16bf;
typedef __attribute__((ext_vector_type(8)))  float  v8f;

static __device__ __forceinline__ unsigned short f2bf_u(float f) {
    union { float f; unsigned u; } in; in.f = f;
    unsigned r = in.u + 0x7fffu + ((in.u >> 16) & 1u);   // round-to-nearest-even
    return (unsigned short)(r >> 16);
}

// ---------------- k0: pack W_q (2048x128 f32) into WMMA B-operand layout, bf16 -----
// pack index t = ((kt*8 + ntile)*32 + lane)*16 + j ;  B layout (16-bit, 32x16):
// lane = h*16 + n%16 (h = K-half), lane's 16 values are K = kt*32 + (j/8)*16 + h*8 + j%8, col n.
__global__ void __launch_bounds__(256)
k0_pack_wq(const float* __restrict__ Wq, unsigned short* __restrict__ pack) {
    int t = blockIdx.x * 256 + threadIdx.x;   // 0 .. 262143
    int j    = t & 15;
    int lane = (t >> 4) & 31;
    int ntl  = (t >> 9) & 7;
    int kt   = t >> 12;
    int h = lane >> 4;
    int n = ntl * 16 + (lane & 15);
    int k = kt * 32 + (j >> 3) * 16 + h * 8 + (j & 7);
    pack[t] = f2bf_u(Wq[k * DQ + n]);
}

// ---------------- k1: fused scores (f32) + queries GEMM (bf16 WMMA) ----------------
__global__ void __launch_bounds__(256)
k1_gemm(const float* __restrict__ x, const unsigned char* __restrict__ mask,
        const float* __restrict__ Wsc, const float* __restrict__ bsc,
        const float* __restrict__ bq, const unsigned short* __restrict__ wqp,
        float* __restrict__ queries, float* __restrict__ scores) {
    __shared__ float sWs[INF_];
    __shared__ __attribute__((aligned(32))) unsigned short sA[1024]; // 2 tiles of 16x32 bf16, A-layout
    __shared__ float sred[256];

    const int tid = threadIdx.x;
    const int blk = blockIdx.x;
    const int b   = blk >> 8;            // grid = B * 256
    const int n0  = (blk & 255) * 32;

    for (int i = tid; i < INF_; i += 256) sWs[i] = Wsc[i];

    // loader mapping: thread owns row = tid/8 (0..31), 4 consecutive k's at kbase
    const int row   = tid >> 3;
    const int kbase = (tid & 7) * 4;
    const float* xrow = x + ((long)b * NN + n0 + row) * RSTRIDE + META;

    // A-register layout (16-bit 16x32): lane = g*16+m, VGPR v: K = (v/4)*16 + (v%4)*2 + e + 8g
    int pidx[4];
#pragma unroll
    for (int i = 0; i < 4; ++i) {
        int kk = kbase + i;
        int g  = (kk & 15) >> 3;
        int v  = (kk >> 4) * 4 + ((kk & 7) >> 1);
        int e  = kk & 1;
        pidx[i] = (row >> 4) * 512 + (g * 16 + (row & 15)) * 16 + 2 * v + e;
    }

    const int wave = tid >> 5;
    const int lane = tid & 31;
    const v16bf* bsrc = (const v16bf*)wqp;
    v8f c0 = {}; v8f c1 = {};
    float sacc = 0.0f;

    __syncthreads();

    for (int it = 0; it < INF_ / 32; ++it) {
        const int k0 = it * 32;
        float a0 = xrow[k0 + kbase + 0];
        float a1 = xrow[k0 + kbase + 1];
        float a2 = xrow[k0 + kbase + 2];
        float a3 = xrow[k0 + kbase + 3];
        float4 wv = *(const float4*)&sWs[k0 + kbase];
        sacc += a0 * wv.x + a1 * wv.y + a2 * wv.z + a3 * wv.w;
        sA[pidx[0]] = f2bf_u(a0);
        sA[pidx[1]] = f2bf_u(a1);
        sA[pidx[2]] = f2bf_u(a2);
        sA[pidx[3]] = f2bf_u(a3);
        __syncthreads();
        v16bf av0 = *(const v16bf*)&sA[lane * 16];
        v16bf av1 = *(const v16bf*)&sA[512 + lane * 16];
        v16bf bv  = bsrc[(it * 8 + wave) * 32 + lane];
        c0 = __builtin_amdgcn_wmma_f32_16x16x32_bf16(false, av0, false, bv, (short)0, c0, false, false);
        c1 = __builtin_amdgcn_wmma_f32_16x16x32_bf16(false, av1, false, bv, (short)0, c1, false, false);
        __syncthreads();
    }

    // D layout: VGPR r -> M = r + 8*(lane/16), N = lane%16 (within this wave's 16 cols)
    const int nq = wave * 16 + (lane & 15);
    const float bqv = bq[nq];
    const int mb = (lane >> 4) * 8;
#pragma unroll
    for (int r = 0; r < 8; ++r) {
        int m = r + mb;
        queries[((long)b * NN + n0 + m) * DQ + nq]      = c0[r] + bqv;
        queries[((long)b * NN + n0 + 16 + m) * DQ + nq] = c1[r] + bqv;
    }

    // reduce 8 partials per row -> masked score
    sred[tid] = sacc;
    __syncthreads();
    if (tid < 32) {
        float s = 0.0f;
#pragma unroll
        for (int i = 0; i < 8; ++i) s += sred[tid * 8 + i];
        s += bsc[0];
        long n = (long)b * NN + n0 + tid;
        if (mask[n]) s = -__builtin_inff();
        scores[n] = s;
    }
}

// ---------------- k2: per-batch max/argmax of scores + gather q_top ---------------
__global__ void __launch_bounds__(256)
k2_argmax(const float* __restrict__ scores, const float* __restrict__ queries,
          int* __restrict__ topidx, float* __restrict__ ntiles, float* __restrict__ qtop) {
    __shared__ float sv[256];
    __shared__ int   si[256];
    const int b = blockIdx.x, t = threadIdx.x;
    const float* s = scores + (long)b * NN;
    float best = -__builtin_inff(); int bi = 0;
    for (int i = 0; i < NN / 256; ++i) {
        int n = t * (NN / 256) + i;           // contiguous chunk => first-max wins
        float v = s[n];
        if (v > best) { best = v; bi = n; }
    }
    sv[t] = best; si[t] = bi;
    __syncthreads();
    for (int off = 128; off > 0; off >>= 1) {
        if (t < off) {
            float ov = sv[t + off]; int oi = si[t + off];
            if (ov > sv[t] || (ov == sv[t] && oi < si[t])) { sv[t] = ov; si[t] = oi; }
        }
        __syncthreads();
    }
    if (t == 0) { topidx[b] = si[0]; ntiles[b] = sv[0]; }
    int idx = si[0];
    if (t < DQ) qtop[b * DQ + t] = queries[((long)b * NN + idx) * DQ + t];
}

// ---------------- k3a: attn_logits = queries . q_top, masked ----------------------
__global__ void __launch_bounds__(256)
k3a_logits(const float* __restrict__ queries, const float* __restrict__ qtop,
           const unsigned char* __restrict__ mask, float* __restrict__ logits) {
    __shared__ __attribute__((aligned(16))) float sq[DQ];
    const int blk = blockIdx.x;               // grid = B * 128
    const int b  = blk >> 7;
    const int n0 = (blk & 127) * 64;
    const int t  = threadIdx.x;
    if (t < DQ) sq[t] = qtop[b * DQ + t];
    __syncthreads();
    const int r = t >> 2, part = t & 3;
    const int n = n0 + r;
    const float4* qp = (const float4*)(queries + ((long)b * NN + n) * DQ + part * 32);
    const float4* tp = (const float4*)(sq + part * 32);
    float acc = 0.f;
#pragma unroll
    for (int i = 0; i < 8; ++i) {
        float4 a = qp[i], c = tp[i];
        acc += a.x * c.x + a.y * c.y + a.z * c.z + a.w * c.w;
    }
    acc += __shfl_xor(acc, 1, 32);
    acc += __shfl_xor(acc, 2, 32);
    if (part == 0) {
        long idx = (long)b * NN + n;
        logits[idx] = mask[idx] ? -__builtin_inff() : acc;
    }
}

// ---------------- k3b: per-batch softmax stats (max, sum); zero pooled ------------
__global__ void __launch_bounds__(256)
k3b_stats(const float* __restrict__ logits, float* __restrict__ stats,
          float* __restrict__ pooled) {
    __shared__ float sred[256];
    const int b = blockIdx.x, t = threadIdx.x;
    const float* l = logits + (long)b * NN;
    float v[32];
    float mx = -__builtin_inff();
#pragma unroll
    for (int i = 0; i < 32; ++i) { v[i] = l[t + i * 256]; mx = fmaxf(mx, v[i]); }
    sred[t] = mx; __syncthreads();
    for (int off = 128; off > 0; off >>= 1) {
        if (t < off) sred[t] = fmaxf(sred[t], sred[t + off]);
        __syncthreads();
    }
    float bmax = sred[0];
    __syncthreads();
    float s = 0.f;
#pragma unroll
    for (int i = 0; i < 32; ++i) s += expf(v[i] - bmax);
    sred[t] = s; __syncthreads();
    for (int off = 128; off > 0; off >>= 1) {
        if (t < off) sred[t] += sred[t + off];
        __syncthreads();
    }
    if (t == 0) { stats[2 * b] = bmax; stats[2 * b + 1] = sred[0]; }
    for (int f = t; f < INF_; f += 256) pooled[b * INF_ + f] = 0.f;
}

// ---------------- k4: pooled = sum_n attn_n * feats_n (second big pass) -----------
__global__ void __launch_bounds__(256)
k4_pool(const float* __restrict__ x, const float* __restrict__ logits,
        const float* __restrict__ stats, float* __restrict__ pooled) {
    __shared__ float sw[1024];
    const int blk = blockIdx.x;               // grid = B*8*8
    const int b  = blk >> 6;
    const int fb = (blk >> 3) & 7;
    const int nb = blk & 7;
    const int t  = threadIdx.x;
    const float bmax = stats[2 * b];
    const float inv  = 1.0f / stats[2 * b + 1];
    const float* l = logits + (long)b * NN + nb * 1024;
#pragma unroll
    for (int i = 0; i < 4; ++i) {
        int j = t + i * 256;
        sw[j] = expf(l[j] - bmax) * inv;      // exp(-inf)=0 handles mask
    }
    __syncthreads();
    const int f = fb * 256 + t;
    const float* xp = x + ((long)b * NN + nb * 1024) * RSTRIDE + META + f;
    float acc = 0.f;
#pragma unroll 4
    for (int j = 0; j < 1024; ++j) acc += sw[j] * xp[(long)j * RSTRIDE];
    atomicAdd(&pooled[b * INF_ + f], acc);
}

// ---------------- k5: bag logit + combine -----------------------------------------
__global__ void __launch_bounds__(256)
k5_final(const float* __restrict__ pooled, const float* __restrict__ Wm,
         const float* __restrict__ bm, const float* __restrict__ ntiles,
         float* __restrict__ out) {
    __shared__ float sred[256];
    const int b = blockIdx.x, t = threadIdx.x;
    float acc = 0.f;
    for (int f = t; f < INF_; f += 256) acc += pooled[b * INF_ + f] * Wm[f];
    sred[t] = acc; __syncthreads();
    for (int off = 128; off > 0; off >>= 1) {
        if (t < off) sred[t] += sred[t + off];
        __syncthreads();
    }
    if (t == 0) out[b] = 0.5f * (ntiles[b] + (sred[0] + bm[0]));
}

extern "C" void kernel_launch(void* const* d_in, const int* in_sizes, int n_in,
                              void* d_out, int out_size, void* d_ws, size_t ws_size,
                              hipStream_t stream) {
    const float*         x    = (const float*)d_in[0];
    const unsigned char* mask = (const unsigned char*)d_in[1];
    const float*         Wsc  = (const float*)d_in[2];
    const float*         bsc  = (const float*)d_in[3];
    const float*         Wq   = (const float*)d_in[4];
    const float*         bq   = (const float*)d_in[5];
    const float*         Wm   = (const float*)d_in[6];
    const float*         bm   = (const float*)d_in[7];
    float* out = (float*)d_out;

    char* ws = (char*)d_ws;
    constexpr size_t WQP = 0;                                 // 512 KB bf16 packed W_q
    constexpr size_t QO  = WQP + 524288;                      // queries: B*N*DQ f32 = 64 MB
    constexpr size_t SCO = QO  + (size_t)BB * NN * DQ * 4;    // scores: 512 KB
    constexpr size_t LGO = SCO + (size_t)BB * NN * 4;         // logits: 512 KB
    constexpr size_t QTO = LGO + (size_t)BB * NN * 4;         // q_top: 8 KB
    constexpr size_t TIO = QTO + BB * DQ * 4;                 // top_idx
    constexpr size_t NTO = TIO + 256;                         // n_tiles_logits
    constexpr size_t STO = NTO + 256;                         // stats (max,sum)
    constexpr size_t PLO = STO + 256;                         // pooled: 128 KB

    unsigned short* wqp  = (unsigned short*)(ws + WQP);
    float* queries = (float*)(ws + QO);
    float* scores  = (float*)(ws + SCO);
    float* logits  = (float*)(ws + LGO);
    float* qtop    = (float*)(ws + QTO);
    int*   topidx  = (int*)(ws + TIO);
    float* ntiles  = (float*)(ws + NTO);
    float* stats   = (float*)(ws + STO);
    float* pooled  = (float*)(ws + PLO);

    k0_pack_wq<<<1024, 256, 0, stream>>>(Wq, wqp);
    k1_gemm<<<BB * (NN / 32), 256, 0, stream>>>(x, mask, Wsc, bsc, bq, wqp, queries, scores);
    k2_argmax<<<BB, 256, 0, stream>>>(scores, queries, topidx, ntiles, qtop);
    k3a_logits<<<BB * (NN / 64), 256, 0, stream>>>(queries, qtop, mask, logits);
    k3b_stats<<<BB, 256, 0, stream>>>(logits, stats, pooled);
    k4_pool<<<BB * 8 * 8, 256, 0, stream>>>(x, logits, stats, pooled);
    k5_final<<<BB, 256, 0, stream>>>(pooled, Wm, bm, ntiles, out);

    (void)in_sizes; (void)n_in; (void)out_size; (void)ws_size;
}
